// GraphUnet_66529043415274
// MI455X (gfx1250) — matrix operand order repaired and checked by reading, not compiled, not run on hardware
//
#include <hip/hip_runtime.h>

// ---------------------------------------------------------------------------
// Graph-U-Net forward on MI455X (gfx1250, wave32).
//
// Precision strategy: the normalized adjacency is a 0/1 mask scaled by
// dinv_i*dinv_j.  0/1 is exact in bf16, so both the dense GEMMs (H@W) and the
// mask-SpMM (M @ X) run as v_wmma_f32_16x16x32_bf16 with f32 accumulate;
// X rows are pre-scaled by dinv_j and result rows scaled by dinv_i in the
// fused epilogue (+bias, ReLU, skip-add).
//
// Data movement: the bf16 mask (6000x6016, zero-padded columns) is 72 MB ->
// resident in the 192 MB L2, so all repeated SpMM reads across the 7 GCN
// layers hit L2, not HBM.  Pooled levels materialize their gathered mask
// (mask0[g[p],g[q]], zero-padded to 16x32 tiles) once per SpMM so the WMMA
// inner loop is always branch-free contiguous global_load_b128 -> issue rate
// is WMMA-bound, not gather-bound.  Falls back to guarded element gathers if
// the workspace is too small for the 58.5 MB tile buffer (decision depends
// only on ws_size -> deterministic / graph-capture safe).
//
// Workspace: ~117 MB base, ~176 MB with the materialized-mask fast path.
// ---------------------------------------------------------------------------

#define DEVINL static __device__ __forceinline__

static constexpr int N0   = 6000;   // input graph
static constexpr int LDM0 = 6016;   // maskB row stride (pad32), tail cols zero
static constexpr int FIN  = 512;
static constexpr int DIM  = 256;
static constexpr int NC   = 7;
static constexpr int N1   = 5400;   // int(0.9*6000)
static constexpr int N2   = 3779;   // int(0.7*5400)  (FP rounding!)
static constexpr int N3   = 2267;   // int(0.6*3779)

typedef __attribute__((ext_vector_type(16))) __bf16        v16bf;
typedef __attribute__((ext_vector_type(8)))  float         v8f;
typedef __attribute__((ext_vector_type(4)))  unsigned int  v4u;

union FragU { v4u q[2]; v16bf v; };   // 32 bytes, trivially copyable

DEVINL __bf16 f2bf(float x) { return (__bf16)x; }

// ======================= element-wise / prep kernels =======================

// maskB[i, j] = (adj!=0 | I), bf16 0/1, row stride LDM0, cols >= N0 zeroed
__global__ void k_build_mask(const float* __restrict__ adj, __bf16* __restrict__ maskB) {
  int t = blockIdx.x * 256 + threadIdx.x;
  if (t >= N0 * LDM0) return;
  int i = t / LDM0, j = t - i * LDM0;
  __bf16 v = f2bf(0.0f);
  if (j < N0) v = (adj[(size_t)i * N0 + j] != 0.0f || i == j) ? f2bf(1.0f) : f2bf(0.0f);
  maskB[t] = v;
}

// pooled mask tile: maskL[p, q] = mask0[g[p], g[q]], zero-padded rows/cols
__global__ void k_gather_mask(const __bf16* __restrict__ maskB, const int* __restrict__ gidx,
                              __bf16* __restrict__ maskL, int n, int mp, int ldm) {
  int t = blockIdx.x * 256 + threadIdx.x;
  if (t >= mp * ldm) return;
  int p = t / ldm, q = t - p * ldm;
  __bf16 v = f2bf(0.0f);
  if (p < n && q < n) v = maskB[(size_t)gidx[p] * LDM0 + gidx[q]];
  maskL[t] = v;
}

// dinv[p] = rsqrt(sum_q mask0[g[p], g[q]])  (gidx==nullptr -> identity)
__global__ void k_degree(const __bf16* __restrict__ maskB, const int* __restrict__ gidx,
                         float* __restrict__ dinv, int n) {
  __shared__ float red[256];
  int p = blockIdx.x;
  int row = gidx ? gidx[p] : p;
  const __bf16* mrow = maskB + (size_t)row * LDM0;
  float s = 0.0f;
  for (int q = threadIdx.x; q < n; q += 256)
    s += (float)mrow[gidx ? gidx[q] : q];
  red[threadIdx.x] = s; __syncthreads();
  for (int st = 128; st > 0; st >>= 1) {
    if (threadIdx.x < st) red[threadIdx.x] += red[threadIdx.x + st];
    __syncthreads();
  }
  if (threadIdx.x == 0) dinv[p] = rsqrtf(red[0]);
}

// WT[c*K + k] = bf16(W[k*C + c])   (transpose so WMMA B-fragments are contiguous)
__global__ void k_wt_bf16(const float* __restrict__ W, __bf16* __restrict__ WT, int K, int C) {
  int t = blockIdx.x * 256 + threadIdx.x;
  if (t >= K * C) return;
  int k = t / C, c = t - k * C;
  WT[(size_t)c * K + k] = f2bf(W[t]);
}

// hA[p,k] = bf16(src[p,k]), zero rows [n, Mpad)
__global__ void k_convertA(const float* __restrict__ src, __bf16* __restrict__ hA,
                           int n, int Mpad, int K) {
  int t = blockIdx.x * 256 + threadIdx.x;
  if (t >= Mpad * K) return;
  int p = t / K, k = t - p * K;
  hA[t] = (p < n) ? f2bf(src[(size_t)p * K + k]) : f2bf(0.0f);
}

// pooled encoder input: hA[p,:] = bf16(src[idx[p],:] * score[idx[p]]), pad rows zero
__global__ void k_gatherScaleA(const float* __restrict__ src, const int* __restrict__ idx,
                               const float* __restrict__ sc, __bf16* __restrict__ hA,
                               int n, int Mpad) {
  int t = blockIdx.x * 256 + threadIdx.x;
  if (t >= Mpad * DIM) return;
  int p = t / DIM, k = t - p * DIM;
  if (p < n) { int ip = idx[p]; hA[t] = f2bf(src[(size_t)ip * DIM + k] * sc[ip]); }
  else hA[t] = f2bf(0.0f);
}

// XsT[c, p] = bf16(XW[p, c] * dinv[p]), zero-pad p >= n
__global__ void k_build_xst(const float* __restrict__ XW, const float* __restrict__ dinv,
                            __bf16* __restrict__ XsT, int n, int npad) {
  int t = blockIdx.x * 256 + threadIdx.x;
  if (t >= DIM * npad) return;
  int c = t / npad, p = t - c * npad;
  XsT[t] = (p < n) ? f2bf(XW[(size_t)p * DIM + c] * dinv[p]) : f2bf(0.0f);
}

__global__ void k_zero_xst(__bf16* __restrict__ XsT, int npad) {
  int t = blockIdx.x * 256 + threadIdx.x;
  if (t < DIM * npad) XsT[t] = f2bf(0.0f);
}

// decoder unpool folded into operand build: XsT[c, idx[p]] = bf16(T[p,c]*dinv[idx[p]])
__global__ void k_scatter_xst(const float* __restrict__ T, const int* __restrict__ idx,
                              const float* __restrict__ dinv, __bf16* __restrict__ XsT,
                              int nsmall, int npad) {
  int t = blockIdx.x * 256 + threadIdx.x;
  if (t >= nsmall * DIM) return;
  int p = t % nsmall, c = t / nsmall;
  int j = idx[p];
  XsT[(size_t)c * npad + j] = f2bf(T[(size_t)p * DIM + c] * dinv[j]);
}

// s[p] = sigmoid(h[p,:].pw + pb)
__global__ void k_scores(const float* __restrict__ h, const float* __restrict__ pw,
                         const float* __restrict__ pb, float* __restrict__ s, int n) {
  int p = blockIdx.x * 256 + threadIdx.x;
  if (p >= n) return;
  float acc = pb[0];
  const float* hp = h + (size_t)p * DIM;
  for (int k = 0; k < DIM; ++k) acc += hp[k] * pw[k];
  s[p] = 1.0f / (1.0f + expf(-acc));
}

// rank-based exact top-k (stable: ties broken by lower index, like lax.top_k)
__global__ void k_topk(const float* __restrict__ s, int n, int k, int* __restrict__ idx) {
  __shared__ int red[256];
  int p = blockIdx.x;
  float sp = s[p];
  int c = 0;
  for (int q = threadIdx.x; q < n; q += 256) {
    float sq = s[q];
    c += (sq > sp) || (sq == sp && q < p);
  }
  red[threadIdx.x] = c; __syncthreads();
  for (int st = 128; st > 0; st >>= 1) {
    if (threadIdx.x < st) red[threadIdx.x] += red[threadIdx.x + st];
    __syncthreads();
  }
  if (threadIdx.x == 0) { int r = red[0]; if (r < k) idx[r] = p; }
}

// gnew[p] = gold ? gold[idx[p]] : idx[p]
__global__ void k_compose(const int* __restrict__ gold, const int* __restrict__ idx,
                          int* __restrict__ gnew, int n) {
  int p = blockIdx.x * 256 + threadIdx.x;
  if (p >= n) return;
  int j = idx[p];
  gnew[p] = gold ? gold[j] : j;
}

// log_softmax(h @ cw + cb)  (NC=7, trivial cost -> plain VALU)
__global__ void k_classifier(const float* __restrict__ h, const float* __restrict__ cw,
                             const float* __restrict__ cb, float* __restrict__ out, int n) {
  int p = blockIdx.x * 256 + threadIdx.x;
  if (p >= n) return;
  float lg[NC];
  #pragma unroll
  for (int c = 0; c < NC; ++c) lg[c] = cb[c];
  const float* hp = h + (size_t)p * DIM;
  for (int k = 0; k < DIM; ++k) {
    float hv = hp[k];
    #pragma unroll
    for (int c = 0; c < NC; ++c) lg[c] += hv * cw[k * NC + c];
  }
  float mx = lg[0];
  #pragma unroll
  for (int c = 1; c < NC; ++c) mx = fmaxf(mx, lg[c]);
  float sum = 0.0f;
  #pragma unroll
  for (int c = 0; c < NC; ++c) sum += expf(lg[c] - mx);
  float lse = mx + logf(sum);
  #pragma unroll
  for (int c = 0; c < NC; ++c) out[(size_t)p * NC + c] = lg[c] - lse;
}

// ============================ WMMA kernels =================================
// Fragment layouts per CDNA5 ISA 7.12.2 (wave32):
//  A (16x32 bf16): lane l -> row m=l&15, half hi=l>>4, b=8*hi;
//     elems [0..7]=A[m, k0+b .. k0+b+7], [8..15]=A[m, k0+16+b .. +7]
//  B (32x16 bf16): lane l -> col c=l&15, elems [0..7]=B[k0+b .. +7, c], [8..15]=B[k0+16+b .. +7, c]
//  C/D (16x16 f32): lane l -> col c=l&15, VGPR v -> row 8*hi+v
// Each wave computes one 16x64 output strip (A frag reused across 4 WMMAs).

#define WMMA_BF16(AC, FA, FB) \
  AC = __builtin_amdgcn_wmma_f32_16x16x32_bf16(false, (FA), false, (FB), (short)0, AC, false, false)

// C[n x 256] = A[Mpad x K] @ W  (B given transposed: BT[c,k]=W[k,c]); K % 32 == 0
__global__ __launch_bounds__(32) void k_gemm_bf16(const __bf16* __restrict__ A,
                                                  const __bf16* __restrict__ BT,
                                                  float* __restrict__ C, int n, int K) {
  const int lane = threadIdx.x;
  const int row0 = blockIdx.x * 16, col0 = blockIdx.y * 64;
  const int m = lane & 15, hi = lane >> 4, b = hi * 8;
  v8f a0 = {}, a1 = {}, a2 = {}, a3 = {};
  const __bf16* ar = A + (size_t)(row0 + m) * K;
  const __bf16* b0 = BT + (size_t)(col0 + m) * K;
  const __bf16* b1 = b0 + (size_t)16 * K;
  const __bf16* b2 = b0 + (size_t)32 * K;
  const __bf16* b3 = b0 + (size_t)48 * K;
  for (int k0 = 0; k0 < K; k0 += 32) {
    FragU fa, f0, f1, f2, f3;
    fa.q[0] = *(const v4u*)(ar + k0 + b);
    fa.q[1] = *(const v4u*)(ar + k0 + 16 + b);
    const int kb = k0 + hi * 8;
    f0.q[0] = *(const v4u*)(b0 + kb); f0.q[1] = *(const v4u*)(b0 + kb + 16);
    f1.q[0] = *(const v4u*)(b1 + kb); f1.q[1] = *(const v4u*)(b1 + kb + 16);
    f2.q[0] = *(const v4u*)(b2 + kb); f2.q[1] = *(const v4u*)(b2 + kb + 16);
    f3.q[0] = *(const v4u*)(b3 + kb); f3.q[1] = *(const v4u*)(b3 + kb + 16);
    WMMA_BF16(a0, fa.v, f0.v);
    WMMA_BF16(a1, fa.v, f1.v);
    WMMA_BF16(a2, fa.v, f2.v);
    WMMA_BF16(a3, fa.v, f3.v);
  }
  #pragma unroll
  for (int v = 0; v < 8; ++v) {
    int row = row0 + hi * 8 + v;
    if (row < n) {
      float* cr = C + (size_t)row * DIM + col0 + m;
      cr[0] = a0[v]; cr[16] = a1[v]; cr[32] = a2[v]; cr[48] = a3[v];
    }
  }
}

// out[p,c] = relu(dinv[p] * sum_k M[p,k] * XsT[c,k] + bias[c]) (+skip)
// mask: row stride ldm (mult of 32).  If gidx==nullptr and nfast==npad the
// whole K loop is the branch-free contiguous b128 path (mask cols zero-padded).
// Otherwise (fallback) guarded element gathers through gidx.
__global__ __launch_bounds__(32) void k_spmm(const __bf16* __restrict__ mask, int ldm,
                                             const int* __restrict__ gidx,
                                             const __bf16* __restrict__ XsT,
                                             const float* __restrict__ dinv,
                                             const float* __restrict__ bias,
                                             const float* __restrict__ skip,
                                             float* __restrict__ out,
                                             int n, int npad, int nfast) {
  const int lane = threadIdx.x;
  const int row0 = blockIdx.x * 16, col0 = blockIdx.y * 64;
  const int m = lane & 15, hi = lane >> 4, b = hi * 8;
  int rp = row0 + m; if (rp >= n && gidx) rp = n - 1;  // clamp only for gather path
  const __bf16* ar = mask + (size_t)(gidx ? gidx[rp] : rp) * ldm;
  const __bf16* b0 = XsT + (size_t)(col0 + m) * npad;
  const __bf16* b1 = b0 + (size_t)16 * npad;
  const __bf16* b2 = b0 + (size_t)32 * npad;
  const __bf16* b3 = b0 + (size_t)48 * npad;
  v8f a0 = {}, a1 = {}, a2 = {}, a3 = {};
  for (int k0 = 0; k0 < npad; k0 += 32) {
    FragU fa, f0, f1, f2, f3;
    if (k0 + 32 <= nfast) {                  // contiguous path (materialized mask)
      fa.q[0] = *(const v4u*)(ar + k0 + b);
      fa.q[1] = *(const v4u*)(ar + k0 + 16 + b);
    } else {                                 // gathered fallback (L2 hits)
      #pragma unroll
      for (int j = 0; j < 8; ++j) {
        int kk = k0 + b + j, kk2 = k0 + 16 + b + j;
        __bf16 v0 = f2bf(0.0f), v1 = f2bf(0.0f);
        if (kk  < n) v0 = ar[gidx ? gidx[kk]  : kk];
        if (kk2 < n) v1 = ar[gidx ? gidx[kk2] : kk2];
        fa.v[j] = v0; fa.v[8 + j] = v1;
      }
    }
    const int kb = k0 + hi * 8;
    f0.q[0] = *(const v4u*)(b0 + kb); f0.q[1] = *(const v4u*)(b0 + kb + 16);
    f1.q[0] = *(const v4u*)(b1 + kb); f1.q[1] = *(const v4u*)(b1 + kb + 16);
    f2.q[0] = *(const v4u*)(b2 + kb); f2.q[1] = *(const v4u*)(b2 + kb + 16);
    f3.q[0] = *(const v4u*)(b3 + kb); f3.q[1] = *(const v4u*)(b3 + kb + 16);
    WMMA_BF16(a0, fa.v, f0.v);
    WMMA_BF16(a1, fa.v, f1.v);
    WMMA_BF16(a2, fa.v, f2.v);
    WMMA_BF16(a3, fa.v, f3.v);
  }
  #pragma unroll
  for (int v = 0; v < 8; ++v) {
    int row = row0 + hi * 8 + v;
    if (row >= n) continue;
    float di = dinv[row];
    size_t ro = (size_t)row * DIM;
    float o0 = fmaxf(di * a0[v] + bias[col0 + m],      0.0f);
    float o1 = fmaxf(di * a1[v] + bias[col0 + 16 + m], 0.0f);
    float o2 = fmaxf(di * a2[v] + bias[col0 + 32 + m], 0.0f);
    float o3 = fmaxf(di * a3[v] + bias[col0 + 48 + m], 0.0f);
    if (skip) {
      o0 += skip[ro + col0 + m];      o1 += skip[ro + col0 + 16 + m];
      o2 += skip[ro + col0 + 32 + m]; o3 += skip[ro + col0 + 48 + m];
    }
    out[ro + col0 + m] = o0;      out[ro + col0 + 16 + m] = o1;
    out[ro + col0 + 32 + m] = o2; out[ro + col0 + 48 + m] = o3;
  }
}

// ============================== host driver ================================

static inline int cdivi(int a, int b) { return (a + b - 1) / b; }
static inline int pad16(int n) { return cdivi(n, 16) * 16; }
static inline int pad32(int n) { return cdivi(n, 32) * 32; }

extern "C" void kernel_launch(void* const* d_in, const int* in_sizes, int n_in,
                              void* d_out, int out_size, void* d_ws, size_t ws_size,
                              hipStream_t stream) {
  // setup_inputs() dict order
  const float* adj = (const float*)d_in[0];
  const float* hin = (const float*)d_in[1];
  const float* dW[3] = {(const float*)d_in[2],  (const float*)d_in[8],  (const float*)d_in[14]};
  const float* dB[3] = {(const float*)d_in[3],  (const float*)d_in[9],  (const float*)d_in[15]};
  const float* uW[3] = {(const float*)d_in[4],  (const float*)d_in[10], (const float*)d_in[16]};
  const float* uB[3] = {(const float*)d_in[5],  (const float*)d_in[11], (const float*)d_in[17]};
  const float* pW[3] = {(const float*)d_in[6],  (const float*)d_in[12], (const float*)d_in[18]};
  const float* pB[3] = {(const float*)d_in[7],  (const float*)d_in[13], (const float*)d_in[19]};
  const float* bw = (const float*)d_in[20];
  const float* bb = (const float*)d_in[21];
  const float* cw = (const float*)d_in[22];
  const float* cb = (const float*)d_in[23];
  float* out = (float*)d_out;

  // ---- workspace carve-out ----
  char* ws = (char*)d_ws; size_t off = 0;
  auto alloc = [&](size_t bytes) -> char* {
    off = (off + 255) & ~(size_t)255;
    char* p = ws + off; off += bytes; return p;
  };
  __bf16* maskB = (__bf16*)alloc((size_t)N0 * LDM0 * 2);          // 72 MB, L2-resident
  __bf16* wtD0  = (__bf16*)alloc((size_t)DIM * FIN * 2);
  __bf16* wtD1  = (__bf16*)alloc((size_t)DIM * DIM * 2);
  __bf16* wtD2  = (__bf16*)alloc((size_t)DIM * DIM * 2);
  __bf16* wtBt  = (__bf16*)alloc((size_t)DIM * DIM * 2);
  __bf16* wtU0  = (__bf16*)alloc((size_t)DIM * DIM * 2);
  __bf16* wtU1  = (__bf16*)alloc((size_t)DIM * DIM * 2);
  __bf16* wtU2  = (__bf16*)alloc((size_t)DIM * DIM * 2);
  __bf16* hA    = (__bf16*)alloc((size_t)pad32(N0) * FIN * 2);
  float*  XW    = (float*) alloc((size_t)N0 * DIM * 4);
  __bf16* XsT   = (__bf16*)alloc((size_t)DIM * pad32(N0) * 2);
  float* down0  = (float*)alloc((size_t)N0 * DIM * 4);
  float* down1  = (float*)alloc((size_t)N1 * DIM * 4);
  float* down2  = (float*)alloc((size_t)N2 * DIM * 4);
  float* hbA    = (float*)alloc((size_t)N0 * DIM * 4);
  float* hbB    = (float*)alloc((size_t)N0 * DIM * 4);
  float* dinv0  = (float*)alloc((size_t)N0 * 4);
  float* dinv1  = (float*)alloc((size_t)N0 * 4);
  float* dinv2  = (float*)alloc((size_t)N0 * 4);
  float* dinv3  = (float*)alloc((size_t)N0 * 4);
  float* s0     = (float*)alloc((size_t)N0 * 4);
  float* s1     = (float*)alloc((size_t)N0 * 4);
  float* s2     = (float*)alloc((size_t)N0 * 4);
  int* idx0     = (int*)alloc((size_t)N1 * 4);
  int* idx1     = (int*)alloc((size_t)N2 * 4);
  int* idx2     = (int*)alloc((size_t)N3 * 4);
  int* g2       = (int*)alloc((size_t)N2 * 4);
  int* g3       = (int*)alloc((size_t)N3 * 4);
  // optional materialized pooled-mask buffer (largest level: 5408x5408 bf16)
  size_t maskLBytes = (size_t)pad16(N1) * pad32(N1) * 2;
  __bf16* maskL = (__bf16*)alloc(maskLBytes);
  const bool useMat = (off <= ws_size);        // pure function of ws_size
  (void)in_sizes; (void)n_in; (void)out_size;

  const dim3 blk1(256), blkW(32);
  auto ew = [](int total) { return dim3((unsigned)cdivi(total, 256)); };

  // ---- one-time (per call) prep ----
  k_build_mask<<<ew(N0 * LDM0), blk1, 0, stream>>>(adj, maskB);
  k_wt_bf16<<<ew(FIN * DIM), blk1, 0, stream>>>(dW[0], wtD0, FIN, DIM);
  k_wt_bf16<<<ew(DIM * DIM), blk1, 0, stream>>>(dW[1], wtD1, DIM, DIM);
  k_wt_bf16<<<ew(DIM * DIM), blk1, 0, stream>>>(dW[2], wtD2, DIM, DIM);
  k_wt_bf16<<<ew(DIM * DIM), blk1, 0, stream>>>(bw,    wtBt, DIM, DIM);
  k_wt_bf16<<<ew(DIM * DIM), blk1, 0, stream>>>(uW[0], wtU0, DIM, DIM);
  k_wt_bf16<<<ew(DIM * DIM), blk1, 0, stream>>>(uW[1], wtU1, DIM, DIM);
  k_wt_bf16<<<ew(DIM * DIM), blk1, 0, stream>>>(uW[2], wtU2, DIM, DIM);
  k_degree<<<dim3(N0), blk1, 0, stream>>>(maskB, nullptr, dinv0, N0);

  // SpMM dispatch: level-0 (gidx==null) always contiguous on padded maskB;
  // pooled levels either materialize the gathered mask (fast) or gather-fallback.
  auto spmm = [&](const int* gidx, const float* dinv, const float* bias,
                  const float* skip, float* dst, int n) {
    int np16 = pad16(n), np32 = pad32(n);
    dim3 grid(np16 / 16, DIM / 64);
    if (!gidx) {
      k_spmm<<<grid, blkW, 0, stream>>>(maskB, LDM0, nullptr, XsT, dinv, bias, skip,
                                        dst, n, np32, np32);
    } else if (useMat) {
      k_gather_mask<<<ew(np16 * np32), blk1, 0, stream>>>(maskB, gidx, maskL, n, np16, np32);
      k_spmm<<<grid, blkW, 0, stream>>>(maskL, np32, nullptr, XsT, dinv, bias, skip,
                                        dst, n, np32, np32);
    } else {
      k_spmm<<<grid, blkW, 0, stream>>>(maskB, LDM0, gidx, XsT, dinv, bias, skip,
                                        dst, n, np32, 0);
    }
  };

  // one GCN (encoder style): hA already prepared, K given
  auto run_gcn = [&](const __bf16* wt, int K, int n, const int* gidx, const float* dinv,
                     const float* bias, const float* skip, float* dst) {
    int np16 = pad16(n), np32 = pad32(n);
    k_gemm_bf16<<<dim3(np16 / 16, DIM / 64), blkW, 0, stream>>>(hA, wt, XW, n, K);
    k_build_xst<<<ew(DIM * np32), blk1, 0, stream>>>(XW, dinv, XsT, n, np32);
    spmm(gidx, dinv, bias, skip, dst, n);
  };

  // ---- encoder level 0 (n=6000, K=512) ----
  k_convertA<<<ew(pad16(N0) * FIN), blk1, 0, stream>>>(hin, hA, N0, pad16(N0), FIN);
  run_gcn(wtD0, FIN, N0, nullptr, dinv0, dB[0], nullptr, down0);
  k_scores<<<ew(N0), blk1, 0, stream>>>(down0, pW[0], pB[0], s0, N0);
  k_topk<<<dim3(N0), blk1, 0, stream>>>(s0, N0, N1, idx0);
  k_degree<<<dim3(N1), blk1, 0, stream>>>(maskB, idx0, dinv1, N1);

  // ---- encoder level 1 (n=5400) ----
  k_gatherScaleA<<<ew(pad16(N1) * DIM), blk1, 0, stream>>>(down0, idx0, s0, hA, N1, pad16(N1));
  run_gcn(wtD1, DIM, N1, idx0, dinv1, dB[1], nullptr, down1);
  k_scores<<<ew(N1), blk1, 0, stream>>>(down1, pW[1], pB[1], s1, N1);
  k_topk<<<dim3(N1), blk1, 0, stream>>>(s1, N1, N2, idx1);
  k_compose<<<ew(N2), blk1, 0, stream>>>(idx0, idx1, g2, N2);
  k_degree<<<dim3(N2), blk1, 0, stream>>>(maskB, g2, dinv2, N2);

  // ---- encoder level 2 (n=3779) ----
  k_gatherScaleA<<<ew(pad16(N2) * DIM), blk1, 0, stream>>>(down1, idx1, s1, hA, N2, pad16(N2));
  run_gcn(wtD2, DIM, N2, g2, dinv2, dB[2], nullptr, down2);
  k_scores<<<ew(N2), blk1, 0, stream>>>(down2, pW[2], pB[2], s2, N2);
  k_topk<<<dim3(N2), blk1, 0, stream>>>(s2, N2, N3, idx2);
  k_compose<<<ew(N3), blk1, 0, stream>>>(g2, idx2, g3, N3);
  k_degree<<<dim3(N3), blk1, 0, stream>>>(maskB, g3, dinv3, N3);

  // ---- bottom (n=2267) ----
  k_gatherScaleA<<<ew(pad16(N3) * DIM), blk1, 0, stream>>>(down2, idx2, s2, hA, N3, pad16(N3));
  run_gcn(wtBt, DIM, N3, g3, dinv3, bb, nullptr, hbA);

  // decoder: h_small (nsmall rows) -> GEMM -> scatter into XsT -> SpMM(n_up)
  auto run_up = [&](const float* hsrc, int nsmall, const __bf16* wt, const int* idx,
                    const int* gidx_up, const float* dinv_up, int nup,
                    const float* bias, const float* skip, float* dst) {
    int sp16 = pad16(nsmall), up32 = pad32(nup);
    k_convertA<<<ew(sp16 * DIM), blk1, 0, stream>>>(hsrc, hA, nsmall, sp16, DIM);
    k_gemm_bf16<<<dim3(sp16 / 16, DIM / 64), blkW, 0, stream>>>(hA, wt, XW, nsmall, DIM);
    k_zero_xst<<<ew(DIM * up32), blk1, 0, stream>>>(XsT, up32);
    k_scatter_xst<<<ew(nsmall * DIM), blk1, 0, stream>>>(XW, idx, dinv_up, XsT, nsmall, up32);
    spmm(gidx_up, dinv_up, bias, skip, dst, nup);
  };

  // ---- decoder ----
  run_up(hbA, N3, wtU0, idx2, g2,      dinv2, N2, uB[0], down2, hbB);   // u=2 -> 3779
  run_up(hbB, N2, wtU1, idx1, idx0,    dinv1, N1, uB[1], down1, hbA);   // u=1 -> 5400
  run_up(hbA, N1, wtU2, idx0, nullptr, dinv0, N0, uB[2], down0, hbB);   // u=0 -> 6000

  // ---- classifier head ----
  k_classifier<<<ew(N0), blk1, 0, stream>>>(hbB, cw, cb, out, N0);
}
